// stackedIndRNN_encoder_58677843198247
// MI455X (gfx1250) — compile-verified, matrix operand order checked by reading, not run on hardware
//
#include <hip/hip_runtime.h>
#include <hip/hip_bf16.h>

// ---------------------------------------------------------------------------
// Problem dims (fixed by the reference)
// ---------------------------------------------------------------------------
#define T_DIM 256
#define B_DIM 64
#define H_DIM 1024
#define L_DIM 6
#define V_DIM 256
#define M_DIM (T_DIM * B_DIM)   // 16384 rows through all GEMMs
#define BN_EPS 1e-5f

// CDNA5 WMMA vector types (wave32; bf16 A/B, f32 accumulate)
typedef __attribute__((ext_vector_type(16))) __bf16 v16bf;
typedef __attribute__((ext_vector_type(8)))  float  v8f;
typedef __attribute__((ext_vector_type(4)))  int    v4i;

#ifndef __has_builtin
#define __has_builtin(x) 0
#endif

// Async global->LDS path (CDNA5 GLOBAL_LOAD_ASYNC_TO_LDS, ASYNCcnt-tracked).
// Builtin signature (from clang diagnostics): (global int4*, shared int4*, Ii, Ii).
#if __has_builtin(__builtin_amdgcn_global_load_async_to_lds_b128) && \
    __has_builtin(__builtin_amdgcn_s_wait_asynccnt)
#define HAVE_ASYNC_LDS 1
#else
#define HAVE_ASYNC_LDS 0
#endif

#define GLOBAL_AS __attribute__((address_space(1)))
#define LDS_AS    __attribute__((address_space(3)))

// round-to-nearest-even f32 -> bf16 (bit pattern)
__device__ __forceinline__ unsigned short f2bf(float f) {
    union { float f; unsigned u; } v; v.f = f;
    unsigned r = v.u + 0x7FFFu + ((v.u >> 16) & 1u);
    return (unsigned short)(r >> 16);
}

// ---------------------------------------------------------------------------
// f32 -> bf16 weight conversion
// ---------------------------------------------------------------------------
__global__ void convert_bf16_kernel(const float* __restrict__ src,
                                    unsigned short* __restrict__ dst, int n) {
    int i = blockIdx.x * blockDim.x + threadIdx.x;
    if (i < n) dst[i] = f2bf(src[i]);
}

// ---------------------------------------------------------------------------
// Embedding gather -> bf16 activation  (i over M*H, H=1024)
// ---------------------------------------------------------------------------
__global__ void embed_kernel(const int* __restrict__ idx,
                             const float* __restrict__ ew,
                             unsigned short* __restrict__ actBF) {
    int i = blockIdx.x * blockDim.x + threadIdx.x;
    int m = i >> 10;          // row in [0, M)
    int h = i & (H_DIM - 1);  // col in [0, H)
    actBF[i] = f2bf(ew[(size_t)idx[m] * H_DIM + h]);
}

// ---------------------------------------------------------------------------
// GEMM: C[M,N] = A[M,K](bf16) x W[N,K](bf16)^T + bias[N]   (f32 accumulate)
// Block: 256 threads = 8 waves, tile 128(M) x 128(N).
// Wave w owns a 16-row strip (mBase = blk*128 + w*16) and all 128 columns:
// 8 accumulators (16x16) -> one A fragment feeds 8 WMMAs per K-step (1:2
// wmma : ds_load_b128 ratio).
// W tile (128x32 bf16 = 8KB) double-buffered in LDS; staged with
// GLOBAL_LOAD_ASYNC_TO_LDS_B128 (ASYNCcnt) when available, overlapping the
// next tile's DMA with the current tile's WMMAs. One barrier per K-step.
// ---------------------------------------------------------------------------
__global__ __launch_bounds__(256) void gemm_bf16_kernel(
    const unsigned short* __restrict__ A,   // [M,K] bf16
    const unsigned short* __restrict__ W,   // [N,K] bf16 (rows are output cols)
    const float* __restrict__ bias,         // [N]
    float* __restrict__ C,                  // [M,N] f32
    int M, int N, int K)
{
    __shared__ unsigned short ldsB[2][128 * 32] __attribute__((aligned(32)));

    const int lane = threadIdx.x & 31;
    const int wid  = threadIdx.x >> 5;              // 0..7 : M strip
    const int mBase  = blockIdx.x * 128 + wid * 16;
    const int nBlock = blockIdx.y * 128;

    const int l16 = lane & 15;
    const int kh  = (lane >> 4) * 16;               // K-half per lane group

    v8f acc[8];
    v8f zero = {0.f, 0.f, 0.f, 0.f, 0.f, 0.f, 0.f, 0.f};
#pragma unroll
    for (int f = 0; f < 8; ++f) acc[f] = zero;

    const unsigned short* aRow = A + (size_t)(mBase + l16) * K;

    // Cooperative staging coordinates: thread moves 32B of the W tile.
    const int sRow  = threadIdx.x >> 1;             // 0..127 (N-rows of tile)
    const int sHalf = (threadIdx.x & 1) * 16;       // K-half of tile
    const unsigned short* wRow = W + (size_t)(nBlock + sRow) * K + sHalf;

    auto stage = [&](int buf, int kc) {
        unsigned short* dst = &ldsB[buf][sRow * 32 + sHalf];
        const unsigned short* src = wRow + kc;
#if HAVE_ASYNC_LDS
        __builtin_amdgcn_global_load_async_to_lds_b128(
            (GLOBAL_AS v4i*)src, (LDS_AS v4i*)dst, 0, 0);
        __builtin_amdgcn_global_load_async_to_lds_b128(
            (GLOBAL_AS v4i*)(src + 8), (LDS_AS v4i*)(dst + 8), 0, 0);
#else
        *(uint4*)dst       = *(const uint4*)src;
        *((uint4*)dst + 1) = *((const uint4*)src + 1);
#endif
    };

    const int nIter = K >> 5;                       // K-steps of 32
    stage(0, 0);

    for (int t = 0; t < nIter; ++t) {
        const int buf = t & 1;
        const int kc  = t << 5;
#if HAVE_ASYNC_LDS
        __builtin_amdgcn_s_wait_asynccnt(0);        // my DMA for tile t done
#endif
        __syncthreads();                            // tile t visible to all

        if (t + 1 < nIter) {
            stage(buf ^ 1, kc + 32);                // overlap DMA with WMMAs
            __builtin_prefetch(aRow + kc + 32, 0, 1);  // global_prefetch_b8
        }

        // A fragment: row (mBase + l16), 16 bf16 along K
        v16bf a = *(const v16bf*)(aRow + kc + kh);

#pragma unroll
        for (int f = 0; f < 8; ++f) {
            const unsigned short* bp = &ldsB[buf][(f * 16 + l16) * 32 + kh];
            v16bf b = *(const v16bf*)bp;
            acc[f] = __builtin_amdgcn_wmma_f32_16x16x32_bf16(
                /*neg_a=*/false, a, /*neg_b=*/false, b,
                /*c_mod=*/(short)0, acc[f],
                /*reuse_a=*/false, /*reuse_b=*/false);
        }
        // No trailing barrier: next iteration's top barrier separates these
        // LDS reads from the overwrite of this buffer two steps ahead.
    }

    // Epilogue: 16x16 f32 D layout -> VGPR r, lane L: M = r + 8*(L>=16), N = L%16
    const int mRow = mBase + (lane >> 4) * 8;
#pragma unroll
    for (int f = 0; f < 8; ++f) {
        int n = nBlock + f * 16 + l16;
        float bv = bias[n];
#pragma unroll
        for (int r = 0; r < 8; ++r) {
            C[(size_t)(mRow + r) * N + n] = acc[f][r] + bv;
        }
    }
}

// ---------------------------------------------------------------------------
// Residual handling: layers 3/5 (l=2,4) add sums[layer-2]; layers 1/3/5
// (l even) refresh the sum buffer with the new s.
// ---------------------------------------------------------------------------
__global__ void residual_kernel(float* __restrict__ di,
                                float* __restrict__ sumBuf,
                                int addRes, int n) {
    int i = blockIdx.x * blockDim.x + threadIdx.x;
    if (i >= n) return;
    float v = di[i];
    if (addRes) { v += sumBuf[i]; di[i] = v; }
    sumBuf[i] = v;
}

// ---------------------------------------------------------------------------
// BatchNorm statistics: per (t,h), mean/rstd over the batch axis (B=64).
// ---------------------------------------------------------------------------
__global__ void bn_stats_kernel(const float* __restrict__ s,   // [T,B,H]
                                float* __restrict__ meanB,     // [T,H]
                                float* __restrict__ rstdB) {   // [T,H]
    int i = blockIdx.x * blockDim.x + threadIdx.x;  // over T*H
    int t = i >> 10;
    int h = i & (H_DIM - 1);
    const float* p = s + (size_t)t * B_DIM * H_DIM + h;
    float sum = 0.f, sq = 0.f;
#pragma unroll 4
    for (int b = 0; b < B_DIM; ++b) {
        float x = p[(size_t)b * H_DIM];
        sum += x; sq += x * x;
    }
    float mean = sum * (1.0f / B_DIM);
    float var  = sq  * (1.0f / B_DIM) - mean * mean;
    meanB[i] = mean;
    rstdB[i] = rsqrtf(var + BN_EPS);
}

// ---------------------------------------------------------------------------
// IndRNN scan: per (b,h), h_t = relu(bn(s_t) + u*h_{t-1}) over T=256.
// Writes next-layer activation in bf16 and final hidden to d_out.
// ---------------------------------------------------------------------------
__global__ void indrnn_kernel(const float* __restrict__ s,      // [T,B,H]
                              const float* __restrict__ meanB,  // [T,H]
                              const float* __restrict__ rstdB,  // [T,H]
                              const float* __restrict__ gamma,  // [H]
                              const float* __restrict__ beta,   // [H]
                              const float* __restrict__ u_hh,   // [H]
                              const float* __restrict__ h0,     // [B,H]
                              unsigned short* __restrict__ actBF, // [T,B,H] bf16
                              float* __restrict__ hOut) {         // [B,H]
    int i = blockIdx.x * blockDim.x + threadIdx.x;  // over B*H
    int b = i >> 10;
    int h = i & (H_DIM - 1);
    float u  = u_hh[h];
    float g  = gamma[h];
    float be = beta[h];
    float hv = h0[i];
    const size_t stride = (size_t)B_DIM * H_DIM;
    const float* sp = s + (size_t)b * H_DIM + h;
    unsigned short* op = actBF + (size_t)b * H_DIM + h;
    for (int t = 0; t < T_DIM; ++t) {
        float x  = sp[(size_t)t * stride];
        int   th = t * H_DIM + h;
        float bn = (x - meanB[th]) * rstdB[th] * g + be;
        hv = fmaxf(bn + u * hv, 0.f);
        op[(size_t)t * stride] = f2bf(hv);
    }
    hOut[i] = hv;
}

// ---------------------------------------------------------------------------
// Launch
// ---------------------------------------------------------------------------
extern "C" void kernel_launch(void* const* d_in, const int* in_sizes, int n_in,
                              void* d_out, int out_size, void* d_ws, size_t ws_size,
                              hipStream_t stream) {
    const int*   input    = (const int*)  d_in[0];  // [T,B]
    const float* hidden   = (const float*)d_in[1];  // [L,B,H]
    const float* embed_w  = (const float*)d_in[2];  // [VOCAB,H]
    const float* di_w     = (const float*)d_in[3];  // [L,H,H]
    const float* di_b     = (const float*)d_in[4];  // [L,H]
    const float* bn_gamma = (const float*)d_in[5];  // [L,H]
    const float* bn_beta  = (const float*)d_in[6];  // [L,H]
    const float* u_hh     = (const float*)d_in[7];  // [L,H]
    const float* fc_w     = (const float*)d_in[8];  // [VOCAB,H]
    const float* fc_b     = (const float*)d_in[9];  // [VOCAB]

    float* logits    = (float*)d_out;                       // [M, VOCAB]
    float* hiddenOut = logits + (size_t)M_DIM * V_DIM;      // [L, B, H]

    // Workspace layout (256B-aligned sections)
    char*  ws  = (char*)d_ws;
    size_t off = 0;
    auto take = [&](size_t bytes) -> char* {
        char* p = ws + off;
        off += (bytes + 255) & ~(size_t)255;
        return p;
    };
    unsigned short* actBF = (unsigned short*)take((size_t)M_DIM * H_DIM * 2);   // 32 MB
    float*          diBuf = (float*)         take((size_t)M_DIM * H_DIM * 4);   // 64 MB
    float*          sumBuf= (float*)         take((size_t)M_DIM * H_DIM * 4);   // 64 MB
    float*          meanB = (float*)         take((size_t)T_DIM * H_DIM * 4);   //  1 MB
    float*          rstdB = (float*)         take((size_t)T_DIM * H_DIM * 4);   //  1 MB
    unsigned short* wdiBF = (unsigned short*)take((size_t)L_DIM * H_DIM * H_DIM * 2); // 12 MB
    unsigned short* fcBF  = (unsigned short*)take((size_t)V_DIM * H_DIM * 2);   // 0.5 MB

    const int threads = 256;

    // Weight conversion to bf16 (deterministic, redone per call)
    convert_bf16_kernel<<<(L_DIM * H_DIM * H_DIM + threads - 1) / threads, threads, 0, stream>>>(
        di_w, wdiBF, L_DIM * H_DIM * H_DIM);
    convert_bf16_kernel<<<(V_DIM * H_DIM + threads - 1) / threads, threads, 0, stream>>>(
        fc_w, fcBF, V_DIM * H_DIM);

    // Embedding gather -> bf16 activation
    embed_kernel<<<(M_DIM * H_DIM) / threads, threads, 0, stream>>>(input, embed_w, actBF);

    const int nElem = M_DIM * H_DIM;
    for (int l = 0; l < L_DIM; ++l) {
        // di = act @ di_w[l]^T + di_b[l]
        dim3 g1(M_DIM / 128, H_DIM / 128);   // 128 x 8 blocks
        gemm_bf16_kernel<<<g1, threads, 0, stream>>>(
            actBF, wdiBF + (size_t)l * H_DIM * H_DIM, di_b + (size_t)l * H_DIM,
            diBuf, M_DIM, H_DIM, H_DIM);

        // Residual bookkeeping: l=0 save; l=2,4 add then save
        if (l == 0 || l == 2 || l == 4) {
            residual_kernel<<<nElem / threads, threads, 0, stream>>>(
                diBuf, sumBuf, (l > 0) ? 1 : 0, nElem);
        }

        // BN stats over batch axis
        bn_stats_kernel<<<(T_DIM * H_DIM) / threads, threads, 0, stream>>>(diBuf, meanB, rstdB);

        // Fused BN + IndRNN scan; emits next-layer bf16 activation + last hidden
        indrnn_kernel<<<(B_DIM * H_DIM) / threads, threads, 0, stream>>>(
            diBuf, meanB, rstdB,
            bn_gamma + (size_t)l * H_DIM, bn_beta + (size_t)l * H_DIM,
            u_hh + (size_t)l * H_DIM, hidden + (size_t)l * B_DIM * H_DIM,
            actBF, hiddenOut + (size_t)l * B_DIM * H_DIM);
    }

    // logits = act @ fc_w^T + fc_b
    dim3 g2(M_DIM / 128, V_DIM / 128);       // 128 x 2 blocks
    gemm_bf16_kernel<<<g2, threads, 0, stream>>>(
        actBF, fcBF, fc_b, logits, M_DIM, V_DIM, H_DIM);
}
